// GRU_20409684590860
// MI455X (gfx1250) — matrix operand order repaired and checked by reading, not compiled, not run on hardware
//
#include <hip/hip_runtime.h>
#include <hip/hip_bf16.h>

// ---------------------------------------------------------------------------
// GRU persistent kernel for MI455X (gfx1250, wave32, WMMA bf16 16x16x32).
// B=64, T=512, IN=256, H=512, L=2 (L read from device scalar).
// Grid: 128 blocks x 96 threads (3 waves). Block -> one 16x16 tile of h.
// Wave 0: r-gate GEMM, Wave 1: u-gate GEMM, Wave 2: n-gate GEMM + fuse/store.
// Weight slices (U: 48KB, W: 24KB bf16) staged ONCE into LDS (bank-conflict-
// free padded pitch) via GLOBAL_LOAD_ASYNC_TO_LDS (ASYNCcnt-tracked).
// x-gates computed once per timestep, reused across both layers.
// Device-wide spin barrier between cells (h ping-pong in workspace).
// ---------------------------------------------------------------------------

typedef unsigned short ushort_t;
typedef __attribute__((ext_vector_type(16))) __bf16 v16bf;
typedef __attribute__((ext_vector_type(8)))  float  v8f;
typedef __attribute__((ext_vector_type(4)))  int    v4i;

#define B_   64
#define T_   512
#define IN_  256
#define H_   512
#define G3_  1536   // 3*H

// LDS tile pitches (elements). 520*2B = 1040B = 260 dwords; 260 % 64 == 4, so
// the 16 B-fragment lanes land on banks {4n..4n+3}: conflict-free b128 reads.
#define U_PITCH 520
#define W_PITCH 264   // 528B = 132 dwords; 132 % 64 == 4

// Workspace layout (bytes)
constexpr size_t OFF_CTRL = 0;                                     // barrier counter
constexpr size_t OFF_W    = 256;                                   // bf16 W [1536][256]
constexpr size_t OFF_U    = OFF_W   + (size_t)G3_ * IN_ * 2;       // bf16 U [1536][512]
constexpr size_t OFF_X    = OFF_U   + (size_t)G3_ * H_  * 2;       // bf16 x [B][T][IN]
constexpr size_t OFF_HBF  = OFF_X   + (size_t)B_ * T_ * IN_ * 2;   // bf16 h ping-pong [2][B][H]
constexpr size_t OFF_HF   = OFF_HBF + (size_t)2 * B_ * H_ * 2;     // f32  h ping-pong [2][B][H]
// total ~19.5 MB

#if defined(__has_builtin)
#if __has_builtin(__builtin_amdgcn_global_load_async_to_lds_b128)
#define HAVE_ASYNC_LDS 1
#endif
#endif

#define AS_GLOBAL __attribute__((address_space(1)))
#define AS_LOCAL  __attribute__((address_space(3)))

union FragBF {
  v16bf  v;
  float4 f4[2];
};

__device__ __forceinline__ ushort_t f2bf(float f) {
  unsigned u = __float_as_uint(f);
  unsigned r = 0x7FFFu + ((u >> 16) & 1u);   // round-to-nearest-even
  return (ushort_t)((u + r) >> 16);
}

// Async 16B global->LDS copy (falls back to synchronous copy if unavailable).
__device__ __forceinline__ void copy16_g2l(const ushort_t* gp, ushort_t* lp) {
#ifdef HAVE_ASYNC_LDS
  __builtin_amdgcn_global_load_async_to_lds_b128(
      (AS_GLOBAL v4i*)(gp), (AS_LOCAL v4i*)(lp), 0, 0);
#else
  *(float4*)lp = *(const float4*)gp;
#endif
}

// A-fragment (16x32 bf16) from global: lanes 0-15 row M=lane, K=k0..k0+7
// (v0-3) and k0+16..k0+23 (v4-7); lanes 16-31 same rows, K offset +8 / +24.
__device__ __forceinline__ v16bf load_a(const ushort_t* base, int row0, size_t ld,
                                        int k0, int lane) {
  const int hw = lane >> 4;
  const int m  = lane & 15;
  const ushort_t* p = base + (size_t)(row0 + m) * ld + (size_t)(k0 + hw * 8);
  FragBF f;
  f.f4[0] = *(const float4*)(p);
  f.f4[1] = *(const float4*)(p + 16);
  return f.v;
}

// B-fragment (32x16 bf16) from an LDS tile holding U^T columns as rows
// (padded pitch). Lanes 0-15: K=k0..k0+7, k0+16..k0+23; lanes 16-31: K offset +8 / +24.
__device__ __forceinline__ v16bf load_bT_lds(const ushort_t* base, int pitch,
                                             int k0, int lane) {
  const int hw = lane >> 4;
  const int n  = lane & 15;
  const ushort_t* p = base + n * pitch + k0 + hw * 8;
  FragBF f;
  f.f4[0] = *(const float4*)(p);
  f.f4[1] = *(const float4*)(p + 16);
  return f.v;
}

__device__ __forceinline__ void grid_barrier(unsigned* counter, unsigned goal) {
  __threadfence();                      // release stores (h, out)
  __syncthreads();
  if (threadIdx.x == 0) {
    __hip_atomic_fetch_add(counter, 1u, __ATOMIC_RELEASE, __HIP_MEMORY_SCOPE_AGENT);
    while (__hip_atomic_load(counter, __ATOMIC_ACQUIRE, __HIP_MEMORY_SCOPE_AGENT) < goal) {
      __builtin_amdgcn_s_sleep(2);
    }
  }
  __syncthreads();
  __threadfence();                      // acquire: invalidate stale lines
}

__global__ void cvt_bf16(const float* __restrict__ src, ushort_t* __restrict__ dst, int n) {
  for (int i = blockIdx.x * blockDim.x + threadIdx.x; i < n;
       i += gridDim.x * blockDim.x)
    dst[i] = f2bf(src[i]);
}

__global__ __launch_bounds__(96, 1) void gru_persistent(
    const float* __restrict__ bW, const float* __restrict__ bU,
    const int* __restrict__ numLayers,
    float* __restrict__ out, unsigned char* __restrict__ ws) {

  const ushort_t* Wbf = (const ushort_t*)(ws + OFF_W);
  const ushort_t* Ubf = (const ushort_t*)(ws + OFF_U);
  const ushort_t* xbf = (const ushort_t*)(ws + OFF_X);
  ushort_t*       hbf = (ushort_t*)(ws + OFF_HBF);
  float*          hf  = (float*)(ws + OFF_HF);
  unsigned*       ctrl = (unsigned*)(ws + OFF_CTRL);

  const int L = numLayers[0];

  const int tid  = threadIdx.x;
  const int lane = tid & 31;
  const int wave = tid >> 5;            // 0=r, 1=u, 2=n
  const int hw   = lane >> 4;
  const int nIdx = lane & 15;           // column within tile (N)

  const int it   = blockIdx.x & 3;      // batch row-tile (0..3)
  const int jt   = blockIdx.x >> 2;     // H col-tile (0..31)
  const int rowB = it * 16;
  const int colH = jt * 16;

  // LDS: time-invariant weight slices + gate exchange tiles (~77 KB)
  __shared__ ushort_t ldsU[48 * U_PITCH];   // 49,920 B
  __shared__ ushort_t ldsW[48 * W_PITCH];   // 25,344 B
  __shared__ float    lds_r[256];
  __shared__ float    lds_u[256];

  // ---- stage U slice: rows {g*512 + colH + r | g<3, r<16}, 512 K each ----
  // 48 rows * 64 chunks(16B) = 3072 chunks; 96 threads -> 32 iterations.
  for (int idx = tid; idx < 48 * (H_ / 8); idx += 96) {
    const int row   = idx >> 6;          // /64
    const int chunk = idx & 63;
    const int gate  = row >> 4;
    const int rloc  = row & 15;
    const ushort_t* gp = Ubf + (size_t)(gate * H_ + colH + rloc) * H_ + chunk * 8;
    ushort_t*       lp = ldsU + row * U_PITCH + chunk * 8;
    copy16_g2l(gp, lp);
  }
  // ---- stage W slice: 48 rows * 32 chunks = 1536 chunks -> 16 iterations ----
  for (int idx = tid; idx < 48 * (IN_ / 8); idx += 96) {
    const int row   = idx >> 5;          // /32
    const int chunk = idx & 31;
    const int gate  = row >> 4;
    const int rloc  = row & 15;
    const ushort_t* gp = Wbf + (size_t)(gate * H_ + colH + rloc) * IN_ + chunk * 8;
    ushort_t*       lp = ldsW + row * W_PITCH + chunk * 8;
    copy16_g2l(gp, lp);
  }
#ifdef HAVE_ASYNC_LDS
#if __has_builtin(__builtin_amdgcn_s_wait_asynccnt)
  __builtin_amdgcn_s_wait_asynccnt(0);
#else
  asm volatile("s_wait_asynccnt 0x0" ::: "memory");
#endif
#endif

  // ---- init h0 = 0 (both ping-pong buffers) ----
  for (int i = blockIdx.x * 96 + tid; i < B_ * H_; i += gridDim.x * 96) {
    hf[i] = 0.f;  hf[B_ * H_ + i] = 0.f;
    hbf[i] = 0;   hbf[B_ * H_ + i] = 0;
  }
  unsigned epoch = 1;
  grid_barrier(ctrl, epoch * gridDim.x);   // also publishes the LDS staging

  // Per-lane biases for this wave's gate & this tile's columns (time-invariant)
  const int gRow0 = wave * H_ + colH;
  const float myBW = bW[gRow0 + nIdx];
  const float myBU = bU[gRow0 + nIdx];

  const ushort_t* myU = ldsU + wave * 16 * U_PITCH;   // this wave's gate slice
  const ushort_t* myW = ldsW + wave * 16 * W_PITCH;

  int cur = 0;
  for (int t = 0; t < T_; ++t) {
    // ---- x-gate GEMM (shared by both layers): xg = xt @ Wg^T ----
    v8f xg = {0.f, 0.f, 0.f, 0.f, 0.f, 0.f, 0.f, 0.f};
    for (int k = 0; k < IN_; k += 32) {
      v16bf a = load_a(xbf + (size_t)t * IN_, rowB, (size_t)T_ * IN_, k, lane);
      v16bf b = load_bT_lds(myW, W_PITCH, k, lane);
      xg = __builtin_amdgcn_wmma_f32_16x16x32_bf16(false, a, false, b,
                                                   (short)0, xg, false, false);
    }

    for (int l = 0; l < L; ++l) {
      // ---- recurrent GEMM: hg = h @ Ug^T (K = 512), B from LDS ----
      const ushort_t* hsrc = hbf + (size_t)cur * B_ * H_;
      v8f hg = {0.f, 0.f, 0.f, 0.f, 0.f, 0.f, 0.f, 0.f};
      for (int k = 0; k < H_; k += 32) {
        v16bf a = load_a(hsrc, rowB, H_, k, lane);
        v16bf b = load_bT_lds(myU, U_PITCH, k, lane);
        hg = __builtin_amdgcn_wmma_f32_16x16x32_bf16(false, a, false, b,
                                                     (short)0, hg, false, false);
      }

      // ---- gate activations: r/u waves publish via LDS ----
      if (wave == 0) {
        #pragma unroll
        for (int i = 0; i < 8; ++i) {
          float p = xg[i] + hg[i] + myBW + myBU;
          lds_r[(i + hw * 8) * 16 + nIdx] = 1.f / (1.f + __expf(-p));
        }
      } else if (wave == 1) {
        #pragma unroll
        for (int i = 0; i < 8; ++i) {
          float p = xg[i] + hg[i] + myBW + myBU;
          lds_u[(i + hw * 8) * 16 + nIdx] = 1.f / (1.f + __expf(-p));
        }
      }
      __syncthreads();

      // ---- n-wave fuses and stores h_new ----
      if (wave == 2) {
        const float* hprev  = hf  + (size_t)cur       * B_ * H_;
        float*       hnextf = hf  + (size_t)(cur ^ 1) * B_ * H_;
        ushort_t*    hnextb = hbf + (size_t)(cur ^ 1) * B_ * H_;
        #pragma unroll
        for (int i = 0; i < 8; ++i) {
          const int M = i + hw * 8;
          const int b = rowB + M;
          const int j = colH + nIdx;
          float r  = lds_r[M * 16 + nIdx];
          float u  = lds_u[M * 16 + nIdx];
          float hn = hg[i] + myBU;           // h@Un^T + b_Un
          float xn = xg[i] + myBW;           // x@Wn^T + b_Wn
          float nn = tanhf(xn + r * hn);
          float hp = hprev[(size_t)b * H_ + j];
          float hnew = u * hp + (1.f - u) * nn;
          hnextf[(size_t)b * H_ + j] = hnew;
          hnextb[(size_t)b * H_ + j] = f2bf(hnew);
          out[(((size_t)t * L + l) * B_ + b) * H_ + j] = hnew;
        }
      }

      // ---- device-wide barrier before next cell reads new h ----
      ++epoch;
      grid_barrier(ctrl, epoch * gridDim.x);
      cur ^= 1;
    }
  }
}

extern "C" void kernel_launch(void* const* d_in, const int* in_sizes, int n_in,
                              void* d_out, int out_size, void* d_ws, size_t ws_size,
                              hipStream_t stream) {
  (void)in_sizes; (void)n_in; (void)out_size; (void)ws_size;
  const float* x  = (const float*)d_in[0];   // [B,T,IN] f32
  const float* W  = (const float*)d_in[1];   // [3H,IN]  f32
  const float* bW = (const float*)d_in[2];   // [3H]
  const float* U  = (const float*)d_in[3];   // [3H,H]
  const float* bU = (const float*)d_in[4];   // [3H]
  const int*   nl = (const int*)d_in[5];     // num_layers (device scalar)
  float* out = (float*)d_out;
  unsigned char* ws = (unsigned char*)d_ws;

  // Reset barrier counter every call (deterministic across graph replays).
  (void)hipMemsetAsync(ws + OFF_CTRL, 0, 256, stream);

  int n;
  n = G3_ * IN_;
  cvt_bf16<<<(n + 255) / 256, 256, 0, stream>>>(W, (ushort_t*)(ws + OFF_W), n);
  n = G3_ * H_;
  cvt_bf16<<<(n + 255) / 256, 256, 0, stream>>>(U, (ushort_t*)(ws + OFF_U), n);
  n = B_ * T_ * IN_;
  cvt_bf16<<<(n + 255) / 256, 256, 0, stream>>>(x, (ushort_t*)(ws + OFF_X), n);

  gru_persistent<<<128, 96, 0, stream>>>(bW, bU, nl, out, ws);
}
